// PrefixTreeDecoder_60730837566103
// MI455X (gfx1250) — matrix operand order repaired and checked by reading, not compiled, not run on hardware
//
#include <hip/hip_runtime.h>
#include <hip/hip_bf16.h>
#include <math.h>

typedef __attribute__((ext_vector_type(16))) _Float16     v16h;
typedef __attribute__((ext_vector_type(8)))  float        v8f;
typedef __attribute__((ext_vector_type(4)))  float        v4f;

#define D_H      128
#define N_NODES  1118480      // 16+256+4096+65536+1048576
#define N_TILES  69905        // N_NODES / 16, each tile = one sibling group

// ---------------------------------------------------------------------------
// Prep: per-level bias vectors, c_root (softplus head), and Wp -> f16 WMMA
// B-fragments in exact per-lane fragment order.
// ---------------------------------------------------------------------------
__global__ __launch_bounds__(128) void prep_kernel(
    const float* __restrict__ z_c, const float* __restrict__ level_table,
    const float* __restrict__ w1r, const float* __restrict__ b1r,
    const float* __restrict__ w2r, const float* __restrict__ b2r,
    const float* __restrict__ wc1, const float* __restrict__ bc1,
    float* __restrict__ bias, _Float16* __restrict__ Bfrag,
    float* __restrict__ c_root_out)
{
  const int j = threadIdx.x;                    // 0..127
  __shared__ float s_z[D_H];
  __shared__ float s_red[D_H];
  s_z[j] = z_c[j];
  __syncthreads();

  // z_part[j] = z_c @ Wz + bc1   (Wz = wc1 rows 0..127)
  float zp = bc1[j];
  for (int i = 0; i < D_H; ++i) zp += s_z[i] * wc1[i * D_H + j];
  const float ws = wc1[384 * D_H + j];          // Ws = last row of wc1

  // bias[lvl] = z_part + level_table[lvl] @ Wl + log1p(16^(5-lvl)) * Ws
  for (int lvl = 1; lvl <= 5; ++lvl) {
    float lp = 0.f;
    for (int i = 0; i < D_H; ++i)
      lp += level_table[lvl * D_H + i] * wc1[(256 + i) * D_H + j];
    const float szl = logf((float)(1u << (4 * (5 - lvl))) + 1.0f); // log1p(16^(5-lvl))
    bias[lvl * D_H + j] = zp + lp + szl * ws;
  }

  // c_root = softplus(relu(z_c @ w1r + b1r) @ w2r + b2r)
  float h = b1r[j];
  for (int i = 0; i < D_H; ++i) h += s_z[i] * w1r[i * D_H + j];
  h = fmaxf(h, 0.f);
  s_red[j] = h * w2r[j];
  __syncthreads();
  for (int off = 64; off > 0; off >>= 1) {
    if (j < off) s_red[j] += s_red[j + off];
    __syncthreads();
  }
  if (j == 0) {
    float x = s_red[0] + b2r[0];
    c_root_out[0] = (x > 20.f) ? x : log1pf(expf(x));
  }

  // Wp (= wc1 rows 128..255) -> f16 B fragments, fragment-order:
  // idx = ((kc*8 + nt)*32 + lane)*16 + i ; element i of lane's v16h:
  //   K = kc*32 + (i<8 ? i : i+8) + (lane>=16 ? 8 : 0), N = nt*16 + (lane&15)
  for (int idx = j; idx < 16384; idx += D_H) {
    const int i    = idx & 15;
    const int lane = (idx >> 4) & 31;
    const int nt   = (idx >> 9) & 7;
    const int kc   = idx >> 12;
    const int K = kc * 32 + ((i < 8) ? i : i + 8) + ((lane & 16) ? 8 : 0);
    const int N = nt * 16 + (lane & 15);
    Bfrag[idx] = (_Float16)wc1[(128 + K) * D_H + N];
  }
}

// ---------------------------------------------------------------------------
// Main kernel: a PAIR of waves per 16-node tile; each wave owns half of N.
//  - each wave explicitly preloads its 16 B fragments (128 VGPRs) ONCE
//  - per tile: 16 NT b128 gathers up front, f32->f16 converts, 16 WMMAs
//  - intra-wave shfl reduction, then pair-combine through 512 B of LDS
//  Register footprint ~240 VGPRs -> ~4 waves/SIMD for gather-latency hiding.
// ---------------------------------------------------------------------------
__global__ __launch_bounds__(256) void logits_kernel(
    const float* __restrict__ prefix, const int* __restrict__ node_pid,
    const float* __restrict__ bias, const _Float16* __restrict__ Bfrag,
    const float* __restrict__ wc2, float* __restrict__ logits)
{
  __shared__ float sPart[8][16];               // per-wave logit partials

  const int lane = threadIdx.x & 31;
  const int wave = threadIdx.x >> 5;           // 0..7 -> 4 pairs
  const int ncol = lane & 15;
  const int lo   = (lane & 16) ? 8 : 0;        // A-frag K sub-block per half
  const int ntg0 = (wave & 1) * 4;             // this wave's N-tile base

  // ---- explicit one-time preload of this wave's 16 B fragments ----
  v16h bf[16];
  #pragma unroll
  for (int kc = 0; kc < 4; ++kc) {
    #pragma unroll
    for (int ntl = 0; ntl < 4; ++ntl) {
      bf[kc * 4 + ntl] =
          *(const v16h*)(Bfrag + ((kc * 8 + ntg0 + ntl) * 32 + lane) * 16);
    }
  }
  float wv[4];
  #pragma unroll
  for (int ntl = 0; ntl < 4; ++ntl) wv[ntl] = wc2[(ntg0 + ntl) * 16 + ncol];

  // each block handles 4 tiles per iteration (4 wave-pairs)
  for (int base = blockIdx.x * 4; base < N_TILES; base += gridDim.x * 4) {
    const int t = base + (wave >> 1);          // wave-uniform
    const bool active = (t < N_TILES);         // wave-uniform

    if (active) {
      int lvl;
      if      (t < 1)    lvl = 1;
      else if (t < 17)   lvl = 2;
      else if (t < 273)  lvl = 3;
      else if (t < 4369) lvl = 4;
      else               lvl = 5;

      // lane (and lane+16) own row M = lane&15 of the A tile
      const long pid = (long)node_pid[t * 16 + ncol];
      const float* __restrict__ row = prefix + pid * (long)D_H;

      // ---- all 16 streaming gathers for this tile issued up front ----
      v4f araw[16];
      #pragma unroll
      for (int kc = 0; kc < 4; ++kc) {
        const int kA = kc * 32 + lo;        // frag elems 0..7  : K=kA..kA+7
        const int kB = kc * 32 + 16 + lo;   // frag elems 8..15 : K=kB..kB+7
        araw[kc * 4 + 0] = __builtin_nontemporal_load((const v4f*)(row + kA));
        araw[kc * 4 + 1] = __builtin_nontemporal_load((const v4f*)(row + kA + 4));
        araw[kc * 4 + 2] = __builtin_nontemporal_load((const v4f*)(row + kB));
        araw[kc * 4 + 3] = __builtin_nontemporal_load((const v4f*)(row + kB + 4));
      }

      // ---- convert to 4 f16 A fragments ----
      v16h af[4];
      #pragma unroll
      for (int kc = 0; kc < 4; ++kc) {
        #pragma unroll
        for (int q = 0; q < 4; ++q) {
          af[kc][q]      = (_Float16)araw[kc * 4 + 0][q];
          af[kc][4 + q]  = (_Float16)araw[kc * 4 + 1][q];
          af[kc][8 + q]  = (_Float16)araw[kc * 4 + 2][q];
          af[kc][12 + q] = (_Float16)araw[kc * 4 + 3][q];
        }
      }

      v8f acc[4];
      #pragma unroll
      for (int ntl = 0; ntl < 4; ++ntl) {
        #pragma unroll
        for (int q = 0; q < 8; ++q) acc[ntl][q] = 0.f;
      }

      // ---- 16 WMMAs against register-resident B ----
      #pragma unroll
      for (int kc = 0; kc < 4; ++kc) {
        #pragma unroll
        for (int ntl = 0; ntl < 4; ++ntl) {
          acc[ntl] = __builtin_amdgcn_wmma_f32_16x16x32_f16(
              false, af[kc], false, bf[kc * 4 + ntl], (short)0, acc[ntl],
              false, false);
        }
      }

      // ---- epilogue: +bias, ReLU, *wc2, reduce over this wave's 64 cols ----
      const float* __restrict__ bl = bias + lvl * D_H;
      float tj[8];
      #pragma unroll
      for (int q = 0; q < 8; ++q) tj[q] = 0.f;
      #pragma unroll
      for (int ntl = 0; ntl < 4; ++ntl) {
        const float b = bl[(ntg0 + ntl) * 16 + ncol];
        const float w = wv[ntl];
        #pragma unroll
        for (int q = 0; q < 8; ++q) {
          float v = acc[ntl][q] + b;
          v = fmaxf(v, 0.f);
          tj[q] += v * w;
        }
      }
      #pragma unroll
      for (int m = 1; m < 16; m <<= 1) {
        #pragma unroll
        for (int q = 0; q < 8; ++q) tj[q] += __shfl_xor(tj[q], m, 32);
      }
      if (ncol == 0) {
        const int mbase = (lane & 16) ? 8 : 0;  // lanes 0/16: rows 0-7 / 8-15
        #pragma unroll
        for (int q = 0; q < 8; ++q) sPart[wave][mbase + q] = tj[q];
      }
    }
    __syncthreads();
    if (active && !(wave & 1) && lane < 16) {
      logits[t * 16 + lane] = sPart[wave][lane] + sPart[wave + 1][lane];
    }
    __syncthreads();
  }
}

// ---------------------------------------------------------------------------
// Per-level softmax over sibling groups of 16 + mass propagation.
// ---------------------------------------------------------------------------
__global__ __launch_bounds__(256) void propagate_kernel(
    const float* __restrict__ logits, int node_off, int n,
    const float* __restrict__ parent, float* __restrict__ out0,
    float* __restrict__ out1)
{
  const int idx = blockIdx.x * blockDim.x + threadIdx.x;
  if (idx >= n) return;
  float l = logits[node_off + idx];
  float mx = l;
  #pragma unroll
  for (int m = 1; m < 16; m <<= 1) mx = fmaxf(mx, __shfl_xor(mx, m, 32));
  float e = expf(l - mx);
  float s = e;
  #pragma unroll
  for (int m = 1; m < 16; m <<= 1) s += __shfl_xor(s, m, 32);
  const float mass = (e / s) * parent[idx >> 4];
  out0[idx] = mass;
  if (out1) out1[idx] = mass;
}

// ---------------------------------------------------------------------------
extern "C" void kernel_launch(void* const* d_in, const int* in_sizes, int n_in,
                              void* d_out, int out_size, void* d_ws, size_t ws_size,
                              hipStream_t stream)
{
  (void)in_sizes; (void)n_in; (void)out_size; (void)ws_size;
  const float* z_c      = (const float*)d_in[0];
  const float* prefix   = (const float*)d_in[1];
  const float* ltab     = (const float*)d_in[2];
  const float* w1r      = (const float*)d_in[3];
  const float* b1r      = (const float*)d_in[4];
  const float* w2r      = (const float*)d_in[5];
  const float* b2r      = (const float*)d_in[6];
  const float* wc1      = (const float*)d_in[7];
  const float* bc1      = (const float*)d_in[8];
  const float* wc2      = (const float*)d_in[9];
  // d_in[10] = bc2: cancels inside softmax, unused
  const int*   node_pid = (const int*)d_in[11];

  float* out = (float*)d_out;
  char*  ws  = (char*)d_ws;
  float*    logits = (float*)ws;                              // N_NODES f32
  float*    bias   = (float*)(ws + 4473920);                  // 6*128 f32
  _Float16* Bfrag  = (_Float16*)(ws + 4473920 + 3072);        // 16384 f16

  float* c_root = out + 1048576;   // single scalar output slot

  prep_kernel<<<1, 128, 0, stream>>>(z_c, ltab, w1r, b1r, w2r, b2r,
                                     wc1, bc1, bias, Bfrag, c_root);
  logits_kernel<<<2048, 256, 0, stream>>>(prefix, node_pid, bias, Bfrag,
                                          wc2, logits);

  const int lvl_n[5]   = {16, 256, 4096, 65536, 1048576};
  const int lvl_off[5] = {0, 16, 272, 4368, 69904};
  float* mseg[6];
  mseg[0] = c_root;                 // parent of level 1
  mseg[1] = out + 1048577;
  mseg[2] = out + 1048593;
  mseg[3] = out + 1048849;
  mseg[4] = out + 1052945;
  mseg[5] = out + 1118481;
  for (int lvl = 1; lvl <= 5; ++lvl) {
    const int n = lvl_n[lvl - 1];
    float* xhat = (lvl == 5) ? out : nullptr;   // x_hat == m5, write both
    propagate_kernel<<<(n + 255) / 256, 256, 0, stream>>>(
        logits, lvl_off[lvl - 1], n, mseg[lvl - 1], mseg[lvl], xhat);
  }
}
